// MDL_X0_64965675319565
// MI455X (gfx1250) — matrix operand order repaired and checked by reading, not compiled, not run on hardware
//
#include <hip/hip_runtime.h>
#include <hip/hip_bf16.h>

// ---------------------------------------------------------------------------
// CDNA5 / gfx1250 implementation of the GNN forward pass.
//  - Dense GEMMs via v_wmma_f32_16x16x32_bf16 (bf16 A/B, f32 accumulate)
//  - Each wave computes a 32x16 tile (2 accumulators) sharing one B fragment;
//    K processed 64 at a time -> 4 WMMAs per staged LDS chunk
//  - Weights pre-packed to bf16x2 once per launch (B fragments = raw loads)
//  - LDS A-chunk layout 16B-aligned so fragments lower to ds_load_b128
//  - Fused bias / fast-tanh / row L2-norm / residual / accumulate epilogues
//  - SpMM via wave-per-edge float4 gathers + global_atomic_add_f32
// ---------------------------------------------------------------------------

typedef __attribute__((ext_vector_type(16))) __bf16        v16bf;
typedef __attribute__((ext_vector_type(8)))  float         v8f;
typedef __attribute__((ext_vector_type(8)))  unsigned int  v8u;

__device__ __forceinline__ unsigned bf16_rne(float x) {
    unsigned u = __float_as_uint(x);
    return (u + 0x7FFFu + ((u >> 16) & 1u)) >> 16;   // round-to-nearest-even
}
__device__ __forceinline__ unsigned pack_bf16x2(float a, float b) {
    return bf16_rne(a) | (bf16_rne(b) << 16);
}
__device__ __forceinline__ float fast_tanh(float x) {
    // tanh(x) = 1 - 2/(e^{2x}+1); single v_exp_f32, correct saturation at +/-1
    float e = __expf(2.f * x);
    return 1.f - 2.f / (e + 1.f);
}

// ---------------------------------------------------------------------------
// Weight pre-pack: fp32 W[K,Dout] -> u32 Wp[K/2,Dout], Wp[p,n] = bf16x2(W[2p,n],W[2p+1,n])
// (For stacked [L,K,D] tensors K_total = L*K works since layer rows are even.)
// ---------------------------------------------------------------------------
__global__ __launch_bounds__(256) void pack_weight_kernel(
    const float* __restrict__ W, unsigned* __restrict__ Wp, int total, int Dout)
{
    int i = blockIdx.x * 256 + threadIdx.x;
    if (i >= total) return;
    int p = i / Dout, n = i - p * Dout;
    size_t i0 = (size_t)(2 * p) * Dout + n;
    Wp[i] = pack_bf16x2(W[i0], W[i0 + Dout]);
}

// ---------------------------------------------------------------------------
// Fused GEMM: out[M,Dout] = epilogue(A[M,K] @ W[K,Dout])
//   epilogue: (+bias) -> (tanh) -> (row l2norm) -> (+residual) -> store
//             and optionally acc_out += result
// Block = 256 threads = 8 wave32s; block tile = 32 rows x 128 cols.
// Each wave owns a 32x16 column strip: two 16x16 WMMA tiles sharing the same
// B fragment (halves weight traffic). K processed 64 at a time (4 WMMAs).
// Requires K % 64 == 0, Dout % 128 == 0 (true for 512/256/128).
// A-chunk LDS row layout: 36 words = [16 pairs | 4 pad | 16 pairs] so every
// fragment read/write group is 16B-aligned (ds_*_b128-able, conflict-free).
// ---------------------------------------------------------------------------
__global__ __launch_bounds__(256) void gemm_fused_kernel(
    const float* __restrict__ A, const unsigned* __restrict__ Wp,
    const float* __restrict__ bias, const float* __restrict__ residual,
    float* __restrict__ out, float* __restrict__ acc_out,
    int M, int K, int Dout, int do_tanh, int do_norm)
{
    __shared__ unsigned lds_a[32 * 36];   // 32 rows x (16+4+16) bf16-pairs
    __shared__ float    lds_o[32 * 133];  // output tile, padded
    __shared__ float    lds_rn[32];       // reciprocal row norms

    const int t    = threadIdx.x;
    const int wave = t >> 5;
    const int lane = t & 31;
    const int g    = lane >> 4;       // lane half-group (K interleave select)
    const int l16  = lane & 15;
    const int row0 = blockIdx.x * 32;
    const int col0 = blockIdx.y * 128;
    const int ncol = col0 + wave * 16 + l16;   // B column owned by this lane

    v8f acc0 = {};   // rows [0,16) of the block tile
    v8f acc1 = {};   // rows [16,32)

    const int sr  = t >> 3;           // staging row (0..31)
    const int grp = t & 7;            // staging group: 4 pairs = 8 floats
    const int sbase = sr * 36 + (grp < 4 ? 4 * grp : 20 + 4 * (grp - 4));

    for (int k0 = 0; k0 < K; k0 += 64) {
        // ---- stage A chunk [32 x 64] as packed bf16 pairs: 8 floats/thread ----
        {
            int gr = row0 + sr;
            float4 q0 = {0.f, 0.f, 0.f, 0.f}, q1 = q0;
            if (gr < M) {
                const float4* p = (const float4*)(A + (size_t)gr * K + (k0 + 8 * grp));
                q0 = p[0];
                q1 = p[1];
                if (k0 + 64 < K) __builtin_prefetch((const float*)p + 64, 0, 1);
            }
            lds_a[sbase + 0] = pack_bf16x2(q0.x, q0.y);
            lds_a[sbase + 1] = pack_bf16x2(q0.z, q0.w);
            lds_a[sbase + 2] = pack_bf16x2(q1.x, q1.y);
            lds_a[sbase + 3] = pack_bf16x2(q1.z, q1.w);
        }
        __syncthreads();

#pragma unroll
        for (int h = 0; h < 2; ++h) {
            // B 32x16 (16-bit), this code's k order (same as A): lanes 0-15 hold K=0..15, lanes 16-31 K=16..31
            v8u bu;
            const int pbase = (k0 >> 1) + h * 16 + g * 8;
#pragma unroll
            for (int v = 0; v < 8; ++v)
                bu[v] = Wp[(size_t)(pbase + v) * Dout + ncol];
            v16bf bf = __builtin_bit_cast(v16bf, bu);

#pragma unroll
            for (int s = 0; s < 2; ++s) {
                // A 16x32 (16-bit): VGPR v holds K pair kkA,kkA+1
                v8u au;
                const int abase = (l16 + s * 16) * 36 + h * 20;
#pragma unroll
                for (int v = 0; v < 8; ++v) {
                    int kkA = g * 16 + 2 * v;
                    au[v] = lds_a[abase + (kkA >> 1)];
                }
                v16bf af = __builtin_bit_cast(v16bf, au);
                if (s == 0)
                    acc0 = __builtin_amdgcn_wmma_f32_16x16x32_bf16(
                               false, af, false, bf, (short)0, acc0, false, false);
                else
                    acc1 = __builtin_amdgcn_wmma_f32_16x16x32_bf16(
                               false, af, false, bf, (short)0, acc1, false, false);
            }
        }
        __syncthreads();
    }

    // ---- epilogue: registers -> LDS tile (bias, tanh applied here) ----
    {
        float bv = bias ? bias[ncol] : 0.f;
#pragma unroll
        for (int v = 0; v < 8; ++v) {
            int   mrow = v + g * 8;             // C/D layout: M = vgpr + 8*(lane>=16)
            float v0 = acc0[v] + bv;
            float v1 = acc1[v] + bv;
            if (do_tanh) { v0 = fast_tanh(v0); v1 = fast_tanh(v1); }
            lds_o[mrow * 133 + wave * 16 + l16]        = v0;
            lds_o[(mrow + 16) * 133 + wave * 16 + l16] = v1;
        }
    }
    __syncthreads();

    if (do_norm) {
        if (t < 32) {
            float s = 0.f;
#pragma unroll 4
            for (int c = 0; c < 128; ++c) {
                float x = lds_o[t * 133 + c];
                s += x * x;
            }
            lds_rn[t] = 1.f / fmaxf(sqrtf(s), 1e-12f);
        }
        __syncthreads();
    }

    // ---- coalesced store (+residual, +acc_out) ----
#pragma unroll
    for (int i = 0; i < 16; ++i) {
        int idx = i * 256 + t;
        int r = idx >> 7, c = idx & 127;
        int gr = row0 + r;
        if (gr < M) {
            size_t gi  = (size_t)gr * Dout + col0 + c;
            float  val = lds_o[r * 133 + c];
            if (do_norm)  val *= lds_rn[r];
            if (residual) val += residual[gi];
            out[gi] = val;
            if (acc_out) acc_out[gi] += val;
        }
    }
}

// ---------------------------------------------------------------------------
// SpMM: out[r,:] += val * in[c,:]  for each edge (r,c,val); D = 128.
// One wave32 per edge, float4 per lane (128 floats), f32 HW atomics.
// Gather table (51 MB) is L2-resident on MI455X (192 MB L2).
// ---------------------------------------------------------------------------
__global__ __launch_bounds__(256) void spmm_atomic_kernel(
    const int* __restrict__ rows, const int* __restrict__ cols,
    const float* __restrict__ vals, const float* __restrict__ in,
    float* __restrict__ out, int E)
{
    int tid  = blockIdx.x * 256 + threadIdx.x;
    int e    = tid >> 5;
    if (e >= E) return;
    int lane = tid & 31;
    int c    = __ldg(cols + e);
    int r    = __ldg(rows + e);
    float v  = __ldg(vals + e);
    float4 m = *((const float4*)(in + (size_t)c * 128) + lane);
    float* dst = out + (size_t)r * 128 + lane * 4;
    unsafeAtomicAdd(dst + 0, v * m.x);
    unsafeAtomicAdd(dst + 1, v * m.y);
    unsafeAtomicAdd(dst + 2, v * m.z);
    unsafeAtomicAdd(dst + 3, v * m.w);
}

__global__ void zero_kernel(float* __restrict__ p, size_t n)
{
    size_t i      = (size_t)blockIdx.x * blockDim.x + threadIdx.x;
    size_t stride = (size_t)gridDim.x * blockDim.x;
    for (; i < n; i += stride) p[i] = 0.f;
}

// ---------------------------------------------------------------------------
extern "C" void kernel_launch(void* const* d_in, const int* in_sizes, int n_in,
                              void* d_out, int out_size, void* d_ws, size_t ws_size,
                              hipStream_t stream)
{
    const float* feat  = (const float*)d_in[0];   // [N,512]
    const int*   srow  = (const int*)  d_in[1];   // [E]
    const int*   scol  = (const int*)  d_in[2];   // [E]
    const float* sval  = (const float*)d_in[3];   // [E]
    const float* Wf0   = (const float*)d_in[4];   // [512,256]
    const float* bf0   = (const float*)d_in[5];   // [256]
    const float* Wf1   = (const float*)d_in[6];   // [256,128]
    const float* bf1   = (const float*)d_in[7];   // [128]
    const float* Wgnn  = (const float*)d_in[8];   // [2,128,128]
    const float* Wemb  = (const float*)d_in[9];   // [128,128]
    const float* bemb  = (const float*)d_in[10];  // [128]
    const float* Wlft  = (const float*)d_in[11];  // [2,128,128]
    const float* blft  = (const float*)d_in[12];  // [2,128]
    const float* Wrgt  = (const float*)d_in[13];  // [2,128,128]
    const float* brgt  = (const float*)d_in[14];  // [2,128]

    const int N = in_sizes[0] / 512;
    const int E = in_sizes[1];
    const int D = 128;

    float* out_emb = (float*)d_out;               // [N,128]
    float* out_lft = out_emb + (size_t)N * D;     // [N,128]
    float* out_rgt = out_lft + (size_t)N * D;     // [N,128]

    // workspace layout (floats); x1 aliases bufA+bufB (x1 dead before they're used)
    float* ws   = (float*)d_ws;
    float* x1   = ws;                             // [N,256]
    float* bufA = ws;                             // [N,128]  SpMM accumulator h
    float* bufB = ws + (size_t)N * 128;           // [N,128]  gnn_out (layer 0)
    float* x2   = ws + (size_t)N * 256;           // [N,128]  reduced features / gnn_out final
    float* acc  = ws + (size_t)N * 384;           // [N,128]  gnn_acc
    float* t0   = ws + (size_t)N * 512;           // [N,128]  MLP intermediate

    // packed bf16x2 weights after the float region (~545 KB)
    unsigned* Wf0p  = (unsigned*)(ws + (size_t)N * 640);   // 256*256 = 65536
    unsigned* Wf1p  = Wf0p  + 65536;                       // 128*128 = 16384
    unsigned* Wgnnp = Wf1p  + 16384;                       // 2*64*128 = 16384
    unsigned* Wembp = Wgnnp + 16384;                       //  64*128 =  8192
    unsigned* Wlftp = Wembp + 8192;                        // 16384
    unsigned* Wrgtp = Wlftp + 16384;                       // 16384

    const dim3 blk(256);
    const int  gx = (N + 31) / 32;                      // 3125
    const int  spmm_blocks = (E * 32 + 255) / 256;      // wave per edge
    const size_t nd = (size_t)N * D;

    // 0) pack all weights to bf16x2 (tiny; once per launch, deterministic)
    pack_weight_kernel<<<(65536 + 255) / 256, blk, 0, stream>>>(Wf0,  Wf0p,  65536, 256);
    pack_weight_kernel<<<(16384 + 255) / 256, blk, 0, stream>>>(Wf1,  Wf1p,  16384, 128);
    pack_weight_kernel<<<(16384 + 255) / 256, blk, 0, stream>>>(Wgnn, Wgnnp, 16384, 128);
    pack_weight_kernel<<<( 8192 + 255) / 256, blk, 0, stream>>>(Wemb, Wembp,  8192, 128);
    pack_weight_kernel<<<(16384 + 255) / 256, blk, 0, stream>>>(Wlft, Wlftp, 16384, 128);
    pack_weight_kernel<<<(16384 + 255) / 256, blk, 0, stream>>>(Wrgt, Wrgtp, 16384, 128);

    // 1) x1 = tanh(feat @ Wf0 + bf0)        [N,512]x[512,256]
    gemm_fused_kernel<<<dim3(gx, 2), blk, 0, stream>>>(
        feat, Wf0p, bf0, nullptr, x1, nullptr, N, 512, 256, 1, 0);
    // 2) x2 = tanh(x1 @ Wf1 + bf1)          [N,256]x[256,128]
    gemm_fused_kernel<<<dim3(gx, 1), blk, 0, stream>>>(
        x1, Wf1p, bf1, nullptr, x2, nullptr, N, 256, 128, 1, 0);
    // 3) gnn_acc = x2
    hipMemcpyAsync(acc, x2, nd * sizeof(float), hipMemcpyDeviceToDevice, stream);

    // 4) GNN layers: h = spmm(in); out = l2norm(tanh(h @ Wgnn[l])); acc += out
    const float* gin = x2;
    float* gouts[2] = { bufB, x2 };
    for (int l = 0; l < 2; ++l) {
        zero_kernel<<<2048, blk, 0, stream>>>(bufA, nd);
        spmm_atomic_kernel<<<spmm_blocks, blk, 0, stream>>>(
            srow, scol, sval, gin, bufA, E);
        gemm_fused_kernel<<<dim3(gx, 1), blk, 0, stream>>>(
            bufA, Wgnnp + (size_t)l * 8192, nullptr, nullptr,
            gouts[l], acc, N, 128, 128, 1, 1);
        gin = gouts[l];
    }
    // gnn_out final == x2

    // 5) emb = l2norm(acc @ Wemb + bemb)   (no tanh)
    gemm_fused_kernel<<<dim3(gx, 1), blk, 0, stream>>>(
        acc, Wembp, bemb, nullptr, out_emb, nullptr, N, 128, 128, 0, 1);

    // 6) left tower: l0 residual, l1 plain
    gemm_fused_kernel<<<dim3(gx, 1), blk, 0, stream>>>(
        x2, Wlftp, blft, x2, t0, nullptr, N, 128, 128, 1, 0);
    gemm_fused_kernel<<<dim3(gx, 1), blk, 0, stream>>>(
        t0, Wlftp + 8192, blft + D, nullptr, out_lft, nullptr, N, 128, 128, 1, 0);

    // 7) right tower
    gemm_fused_kernel<<<dim3(gx, 1), blk, 0, stream>>>(
        x2, Wrgtp, brgt, x2, t0, nullptr, N, 128, 128, 1, 0);
    gemm_fused_kernel<<<dim3(gx, 1), blk, 0, stream>>>(
        t0, Wrgtp + 8192, brgt + D, nullptr, out_rgt, nullptr, N, 128, 128, 1, 0);
}
